// MultiHeadAttentionWithGating_23390391894206
// MI455X (gfx1250) — matrix operand order repaired and checked by reading, not compiled, not run on hardware
//
#include <hip/hip_runtime.h>
#include <hip/hip_bf16.h>

// ---------------------------------------------------------------------------
// MI455X (gfx1250): bf16 WMMA (v_wmma_f32_16x16x32_bf16) GEMMs, f32 accum.
// Block = 256 threads = 8 wave32 waves (4 x 2); wave tile = 64x64
// (4x4 WMMA fragments, 16 WMMA / K-step); block tile = 256x128.
// K-loop is software-pipelined (double-buffered fragments) so global_load_b128
// latency overlaps the WMMA burst instead of serializing on s_wait_loadcnt 0.
// ---------------------------------------------------------------------------

#define BB 2
#define SS 2048
#define DD 2048
#define HH 16
#define HD 128

typedef __attribute__((ext_vector_type(16))) __bf16 v16bf;
typedef __attribute__((ext_vector_type(8)))  __bf16 v8bf;
typedef __attribute__((ext_vector_type(4)))  __bf16 v4bf;
typedef __attribute__((ext_vector_type(8)))  float  v8f;

#define WMMA_BF16(A_, B_, C_) \
  __builtin_amdgcn_wmma_f32_16x16x32_bf16(false, (A_), false, (B_), (short)0, (C_), false, false)

// ---- fragment loaders (ISA 7.12.2 layouts) --------------------------------
// A 16x32 bf16 (MxK): lanes 0-15 -> M=lane, K=[k0..k0+7] & [k0+16..k0+23]
//                     lanes 16-31 -> M=lane-16, K=[k0+8..k0+15] & [k0+24..k0+31]
__device__ __forceinline__ v16bf load_a_bf16(const __bf16* __restrict__ A, int lda,
                                             int row0, int k0, int lane) {
  int m  = row0 + (lane & 15);
  int kb = k0 + ((lane >> 4) << 3);
  const __bf16* p = A + (size_t)m * lda + kb;
  v8bf lo = *(const v8bf*)p;
  v8bf hi = *(const v8bf*)(p + 16);
  v16bf f;
#pragma unroll
  for (int i = 0; i < 8; ++i) { f[i] = lo[i]; f[i + 8] = hi[i]; }
  return f;
}

// Same A layout but source is f32, converted to bf16 on the fly.
__device__ __forceinline__ v16bf load_a_f32(const float* __restrict__ A, int lda,
                                            int row0, int k0, int lane) {
  int m  = row0 + (lane & 15);
  int kb = k0 + ((lane >> 4) << 3);
  const float* p = A + (size_t)m * lda + kb;
  v8f lo = *(const v8f*)p;
  v8f hi = *(const v8f*)(p + 16);
  v16bf f;
#pragma unroll
  for (int i = 0; i < 8; ++i) { f[i] = (__bf16)lo[i]; f[i + 8] = (__bf16)hi[i]; }
  return f;
}

// B 32x16 bf16 (KxN), B[k][n] = W[n][k] (W row-major NxK):
// lane -> N = col0 + lane%16, K = [k0+8*(lane/16) .. +7] & [k0+16+8*(lane/16) .. +7] in W row.
__device__ __forceinline__ v16bf load_b_bf16(const __bf16* __restrict__ W, int ldb,
                                             int col0, int k0, int lane) {
  int n  = col0 + (lane & 15);
  int kk = k0 + ((lane >> 4) << 3);
  const __bf16* p = W + (size_t)n * ldb + kk;
  v8bf lo = *(const v8bf*)p;
  v8bf hi = *(const v8bf*)(p + 16);
  v16bf f;
#pragma unroll
  for (int i = 0; i < 8; ++i) { f[i] = lo[i]; f[i + 8] = hi[i]; }
  return f;
}

// ---- pipelined 64x64 GEMM core --------------------------------------------
struct Frag { v16bf a[4]; v16bf b[4]; };

template <typename LA, typename LB>
__device__ __forceinline__ void load_frags(Frag& f, const LA& la, const LB& lb, int k0) {
#pragma unroll
  for (int i = 0; i < 4; ++i) f.a[i] = la(i * 16, k0);
#pragma unroll
  for (int j = 0; j < 4; ++j) f.b[j] = lb(j * 16, k0);
}

__device__ __forceinline__ void mma_all(v8f (&acc)[4][4], const Frag& f) {
#pragma unroll
  for (int i = 0; i < 4; ++i)
#pragma unroll
    for (int j = 0; j < 4; ++j)
      acc[i][j] = WMMA_BF16(f.a[i], f.b[j], acc[i][j]);
}

template <typename LA, typename LB>
__device__ __forceinline__ void gemm_pipe(v8f (&acc)[4][4], const LA& la, const LB& lb, int K) {
  Frag cur;
  load_frags(cur, la, lb, 0);
#pragma unroll 2
  for (int k0 = 32; k0 < K; k0 += 32) {
    Frag nxt;
    load_frags(nxt, la, lb, k0);  // issue loads for next slab
    mma_all(acc, cur);            // 16 WMMAs overlap the loads above
    cur = nxt;
  }
  mma_all(acc, cur);
}

// ---- fp32 -> bf16 conversion ----------------------------------------------
__global__ void __launch_bounds__(256) k_cvt(const float* __restrict__ in,
                                             __bf16* __restrict__ out) {
  size_t i = ((size_t)blockIdx.x * 256 + threadIdx.x) * 4;
  float4 v = *(const float4*)(in + i);
  v4bf o;
  o[0] = (__bf16)v.x; o[1] = (__bf16)v.y; o[2] = (__bf16)v.z; o[3] = (__bf16)v.w;
  *(v4bf*)(out + i) = o;
}

// ---- QKV projection: Out = X @ W^T + b, write bf16 in attention layout ----
// mode 0: head-major [B,H,S,HD] (Q with scale, K), mode 1: transposed [B,H,HD,S] (V)
__global__ void __launch_bounds__(256)
k_proj_gemm(const __bf16* __restrict__ X, const __bf16* __restrict__ W,
            const float* __restrict__ bias, __bf16* __restrict__ Out,
            int mode, float scale) {
  int lane = threadIdx.x & 31, wid = threadIdx.x >> 5;
  int row_w = blockIdx.y * 256 + (wid & 3) * 64;
  int col_w = blockIdx.x * 128 + (wid >> 2) * 64;
  v8f acc[4][4] = {};
  auto la = [&](int dr, int k0) {
    if (dr == 0)  // keep next K-slab of A hot (global_prefetch_b8)
      __builtin_prefetch((const void*)(X + (size_t)(row_w + (lane & 15)) * DD + k0 + 64), 0, 3);
    return load_a_bf16(X, DD, row_w + dr, k0, lane);
  };
  auto lb = [&](int dc, int k0) { return load_b_bf16(W, DD, col_w + dc, k0, lane); };
  gemm_pipe(acc, la, lb, DD);

  int hi = (lane >> 4) << 3;
#pragma unroll
  for (int i = 0; i < 4; ++i)
#pragma unroll
    for (int j = 0; j < 4; ++j) {
      int n = col_w + j * 16 + (lane & 15);
      float bv = bias[n];
      int h = n >> 7, d = n & 127;
#pragma unroll
      for (int r = 0; r < 8; ++r) {
        int m = row_w + i * 16 + hi + r;
        int b = m >> 11, s = m & 2047;
        float val = (acc[i][j][r] + bv) * scale;
        size_t idx = (mode == 0)
          ? ((((size_t)b * HH + h) * SS + s) * HD + d)
          : ((((size_t)b * HH + h) * HD + d) * SS + s);
        Out[idx] = (__bf16)val;
      }
    }
}

// ---- scores = Q @ K^T (per head), apply mask, write fp32 into attn buffer --
__global__ void __launch_bounds__(256)
k_scores(const __bf16* __restrict__ Qb, const __bf16* __restrict__ Kb,
         const int* __restrict__ mask, float* __restrict__ attn) {
  int bh = blockIdx.z;
  const __bf16* Qh = Qb + (size_t)bh * SS * HD;
  const __bf16* Kh = Kb + (size_t)bh * SS * HD;
  float* out = attn + (size_t)bh * SS * SS;
  const int* mk = mask + (bh >> 4) * SS;
  int lane = threadIdx.x & 31, wid = threadIdx.x >> 5;
  int row_w = blockIdx.y * 256 + (wid & 3) * 64;
  int col_w = blockIdx.x * 128 + (wid >> 2) * 64;
  v8f acc[4][4] = {};
  auto la = [&](int dr, int k0) { return load_a_bf16(Qh, HD, row_w + dr, k0, lane); };
  auto lb = [&](int dc, int k0) { return load_b_bf16(Kh, HD, col_w + dc, k0, lane); };
  gemm_pipe(acc, la, lb, HD);

  int hi = (lane >> 4) << 3;
#pragma unroll
  for (int i = 0; i < 4; ++i)
#pragma unroll
    for (int j = 0; j < 4; ++j) {
      int n = col_w + j * 16 + (lane & 15);
      float keep = (mk[n] != 0) ? 1.0f : 0.0f;
#pragma unroll
      for (int r = 0; r < 8; ++r) {
        int m = row_w + i * 16 + hi + r;
        out[(size_t)m * SS + n] = keep * acc[i][j][r] + (1.0f - keep) * (-1e9f);
      }
    }
}

// ---- row softmax in place (this buffer is also the attn output) -----------
__global__ void __launch_bounds__(256) k_softmax(float* __restrict__ attn) {
  float* p = attn + (size_t)blockIdx.x * SS;
  __shared__ float red[256];
  int t = threadIdx.x;
  float mx = -3.4e38f;
#pragma unroll
  for (int i = 0; i < 8; ++i) mx = fmaxf(mx, p[i * 256 + t]);
  red[t] = mx; __syncthreads();
  for (int s = 128; s > 0; s >>= 1) { if (t < s) red[t] = fmaxf(red[t], red[t + s]); __syncthreads(); }
  mx = red[0]; __syncthreads();
  float vals[8], sum = 0.0f;
#pragma unroll
  for (int i = 0; i < 8; ++i) { vals[i] = __expf(p[i * 256 + t] - mx); sum += vals[i]; }
  red[t] = sum; __syncthreads();
  for (int s = 128; s > 0; s >>= 1) { if (t < s) red[t] += red[t + s]; __syncthreads(); }
  float inv = 1.0f / red[0];
#pragma unroll
  for (int i = 0; i < 8; ++i) p[i * 256 + t] = vals[i] * inv;
}

// ---- ctx = attn @ V (per head); A f32->bf16 on the fly, B = V^T rows ------
__global__ void __launch_bounds__(256)
k_ctx(const float* __restrict__ attn, const __bf16* __restrict__ Vt,
      __bf16* __restrict__ Ctxb) {
  int bh = blockIdx.z, b = bh >> 4, h = bh & 15;
  const float* Ah = attn + (size_t)bh * SS * SS;
  const __bf16* Bh = Vt + (size_t)bh * HD * SS;
  int lane = threadIdx.x & 31, wid = threadIdx.x >> 5;
  int row_w = blockIdx.y * 256 + (wid & 3) * 64;
  int col_w = (wid >> 2) * 64;  // N = HD = 128 total
  v8f acc[4][4] = {};
  auto la = [&](int dr, int k0) { return load_a_f32(Ah, SS, row_w + dr, k0, lane); };
  auto lb = [&](int dc, int k0) { return load_b_bf16(Bh, SS, col_w + dc, k0, lane); };
  gemm_pipe(acc, la, lb, SS);

  int hi = (lane >> 4) << 3;
#pragma unroll
  for (int i = 0; i < 4; ++i)
#pragma unroll
    for (int j = 0; j < 4; ++j) {
      int n = col_w + j * 16 + (lane & 15);
#pragma unroll
      for (int r = 0; r < 8; ++r) {
        int m = row_w + i * 16 + hi + r;
        Ctxb[((size_t)(b * SS + m)) * DD + h * HD + n] = (__bf16)acc[i][j][r];
      }
    }
}

// ---- out = ctx @ Wo^T + bo (fp32 result kept for the gate) ----------------
__global__ void __launch_bounds__(256)
k_out_proj(const __bf16* __restrict__ Ctxb, const __bf16* __restrict__ Wob,
           const float* __restrict__ bo, float* __restrict__ Outp) {
  int lane = threadIdx.x & 31, wid = threadIdx.x >> 5;
  int row_w = blockIdx.y * 256 + (wid & 3) * 64;
  int col_w = blockIdx.x * 128 + (wid >> 2) * 64;
  v8f acc[4][4] = {};
  auto la = [&](int dr, int k0) { return load_a_bf16(Ctxb, DD, row_w + dr, k0, lane); };
  auto lb = [&](int dc, int k0) { return load_b_bf16(Wob, DD, col_w + dc, k0, lane); };
  gemm_pipe(acc, la, lb, DD);

  int hi = (lane >> 4) << 3;
#pragma unroll
  for (int i = 0; i < 4; ++i)
#pragma unroll
    for (int j = 0; j < 4; ++j) {
      int n = col_w + j * 16 + (lane & 15);
      float bv = bo[n];
#pragma unroll
      for (int r = 0; r < 8; ++r) {
        int m = row_w + i * 16 + hi + r;
        Outp[(size_t)m * DD + n] = acc[i][j][r] + bv;
      }
    }
}

// ---- gate GEMM: z = [out, rc] @ Wg^T + bg; Y = sig(z)*out+(1-sig)*rc+hid --
__global__ void __launch_bounds__(256)
k_gate(const float* __restrict__ Outp, const float* __restrict__ rc,
       const float* __restrict__ hidden, const __bf16* __restrict__ Wgb,
       const float* __restrict__ bg, float* __restrict__ Y) {
  int lane = threadIdx.x & 31, wid = threadIdx.x >> 5;
  int row_w = blockIdx.y * 256 + (wid & 3) * 64;
  int col_w = blockIdx.x * 128 + (wid >> 2) * 64;
  v8f acc[4][4] = {};
  auto la = [&](int dr, int k0) {
    const float* src = (k0 < DD) ? Outp : rc;   // uniform branch, EXEC unchanged
    int ks = (k0 < DD) ? k0 : (k0 - DD);
    return load_a_f32(src, DD, row_w + dr, ks, lane);
  };
  auto lb = [&](int dc, int k0) { return load_b_bf16(Wgb, 2 * DD, col_w + dc, k0, lane); };
  gemm_pipe(acc, la, lb, 2 * DD);

  int hi = (lane >> 4) << 3;
#pragma unroll
  for (int i = 0; i < 4; ++i)
#pragma unroll
    for (int j = 0; j < 4; ++j) {
      int n = col_w + j * 16 + (lane & 15);
      float bv = bg[n];
#pragma unroll
      for (int r = 0; r < 8; ++r) {
        int m = row_w + i * 16 + hi + r;
        size_t idx = (size_t)m * DD + n;
        float g = 1.0f / (1.0f + __expf(-(acc[i][j][r] + bv)));
        float o = Outp[idx], rr = rc[idx];
        Y[idx] = g * o + (1.0f - g) * rr + hidden[idx];
      }
    }
}

// ---- LayerNorm over last dim, write final output --------------------------
__global__ void __launch_bounds__(256)
k_layernorm(const float* __restrict__ Y, const float* __restrict__ w,
            const float* __restrict__ b, float* __restrict__ out) {
  const float* p = Y + (size_t)blockIdx.x * DD;
  float* o = out + (size_t)blockIdx.x * DD;
  __shared__ float rs[256], rq[256];
  int t = threadIdx.x;
  float sum = 0.0f, sq = 0.0f, x[8];
#pragma unroll
  for (int i = 0; i < 8; ++i) { x[i] = p[i * 256 + t]; sum += x[i]; sq += x[i] * x[i]; }
  rs[t] = sum; rq[t] = sq; __syncthreads();
  for (int s = 128; s > 0; s >>= 1) {
    if (t < s) { rs[t] += rs[t + s]; rq[t] += rq[t + s]; }
    __syncthreads();
  }
  float u = rs[0] * (1.0f / DD);
  float var = rq[0] * (1.0f / DD) - u * u;
  float rstd = rsqrtf(var + 1e-12f);
#pragma unroll
  for (int i = 0; i < 8; ++i) {
    int c = i * 256 + t;
    o[c] = w[c] * ((x[i] - u) * rstd) + b[c];
  }
}

// ---------------------------------------------------------------------------
extern "C" void kernel_launch(void* const* d_in, const int* in_sizes, int n_in,
                              void* d_out, int out_size, void* d_ws, size_t ws_size,
                              hipStream_t stream) {
  (void)in_sizes; (void)n_in; (void)out_size; (void)ws_size;
  const float* hidden = (const float*)d_in[0];
  const float* rc     = (const float*)d_in[1];
  const int*   mask   = (const int*)d_in[2];
  const float* Wq = (const float*)d_in[3];  const float* bq = (const float*)d_in[4];
  const float* Wk = (const float*)d_in[5];  const float* bk = (const float*)d_in[6];
  const float* Wv = (const float*)d_in[7];  const float* bv = (const float*)d_in[8];
  const float* Wo = (const float*)d_in[9];  const float* bo = (const float*)d_in[10];
  const float* Wg = (const float*)d_in[11]; const float* bg = (const float*)d_in[12];
  const float* lnw = (const float*)d_in[13]; const float* lnb = (const float*)d_in[14];

  float* outMain = (float*)d_out;                               // [B,S,D]
  float* outAttn = (float*)d_out + (size_t)BB * SS * DD;        // [B,H,S,S]

  // workspace carve-up (≈192 MiB)
  char* wsp = (char*)d_ws;
  __bf16* Xb   = (__bf16*)wsp; wsp += (size_t)BB * SS * DD * 2;
  __bf16* Wqb  = (__bf16*)wsp; wsp += (size_t)DD * DD * 2;
  __bf16* Wkb  = (__bf16*)wsp; wsp += (size_t)DD * DD * 2;
  __bf16* Wvb  = (__bf16*)wsp; wsp += (size_t)DD * DD * 2;
  __bf16* Wob  = (__bf16*)wsp; wsp += (size_t)DD * DD * 2;
  __bf16* Wgb  = (__bf16*)wsp; wsp += (size_t)DD * 2 * DD * 2;
  __bf16* Qb   = (__bf16*)wsp; wsp += (size_t)BB * SS * DD * 2;
  __bf16* Kb   = (__bf16*)wsp; wsp += (size_t)BB * SS * DD * 2;
  __bf16* Vt   = (__bf16*)wsp; wsp += (size_t)BB * SS * DD * 2;
  __bf16* Ctxb = (__bf16*)wsp; wsp += (size_t)BB * SS * DD * 2;
  float*  Outp = (float*)wsp;  wsp += (size_t)BB * SS * DD * 4;
  float*  Yb   = (float*)wsp;

  dim3 blk(256);
  // 1) fp32 -> bf16 conversions
  k_cvt<<<dim3((BB * SS * DD) / 1024), blk, 0, stream>>>(hidden, Xb);
  k_cvt<<<dim3((DD * DD) / 1024), blk, 0, stream>>>(Wq, Wqb);
  k_cvt<<<dim3((DD * DD) / 1024), blk, 0, stream>>>(Wk, Wkb);
  k_cvt<<<dim3((DD * DD) / 1024), blk, 0, stream>>>(Wv, Wvb);
  k_cvt<<<dim3((DD * DD) / 1024), blk, 0, stream>>>(Wo, Wob);
  k_cvt<<<dim3((DD * 2 * DD) / 1024), blk, 0, stream>>>(Wg, Wgb);

  // 2) Q/K/V projections (scale folded into Q)
  dim3 gProj(DD / 128, (BB * SS) / 256);
  k_proj_gemm<<<gProj, blk, 0, stream>>>(Xb, Wqb, bq, Qb, 0, 0.08838834764831845f);
  k_proj_gemm<<<gProj, blk, 0, stream>>>(Xb, Wkb, bk, Kb, 0, 1.0f);
  k_proj_gemm<<<gProj, blk, 0, stream>>>(Xb, Wvb, bv, Vt, 1, 1.0f);

  // 3) scores + mask, 4) softmax (writes attn output), 5) ctx = attn @ V
  k_scores<<<dim3(SS / 128, SS / 256, BB * HH), blk, 0, stream>>>(Qb, Kb, mask, outAttn);
  k_softmax<<<dim3(BB * HH * SS), blk, 0, stream>>>(outAttn);
  k_ctx<<<dim3(1, SS / 256, BB * HH), blk, 0, stream>>>(outAttn, Vt, Ctxb);

  // 6) output projection, 7) gate + residual, 8) LayerNorm
  dim3 gD(DD / 128, (BB * SS) / 256);
  k_out_proj<<<gD, blk, 0, stream>>>(Ctxb, Wob, bo, Outp);
  k_gate<<<gD, blk, 0, stream>>>(Outp, rc, hidden, Wgb, bg, Yb);
  k_layernorm<<<dim3(BB * SS), blk, 0, stream>>>(Yb, lnw, lnb, outMain);
}
